// RelationTransformerHead_22153441313074
// MI455X (gfx1250) — matrix operand order repaired and checked by reading, not compiled, not run on hardware
//
#include <hip/hip_runtime.h>
#include <hip/hip_bf16.h>

typedef _Float16 v16h __attribute__((ext_vector_type(16)));
typedef _Float16 v8h  __attribute__((ext_vector_type(8)));
typedef float    v8f  __attribute__((ext_vector_type(8)));

#define NNODES 384
#define DMODEL 256
#define NPAIRS (NNODES * (NNODES - 1))   // 147072
#define SBSTRIDE 520                      // halves; 1040B % 256B = 16 -> conflict-free b128

// ---------------------------------------------------------------------------
// Generic fp32 GEMM: C[m,c] = sum_k A[m,k]*B[k,c] + bias[c]  (optional relu)
// ---------------------------------------------------------------------------
__global__ void gemm_kernel(const float* __restrict__ A,
                            const float* __restrict__ B,
                            const float* __restrict__ bias,
                            float* __restrict__ C,
                            int M, int K, int Nn, int relu) {
    int c = blockIdx.x * blockDim.x + threadIdx.x;
    int m = blockIdx.y;
    if (c >= Nn || m >= M) return;
    float acc = bias ? bias[c] : 0.f;
    const float* ar = A + (long)m * K;
    for (int k = 0; k < K; ++k)
        acc = fmaf(ar[k], B[(long)k * Nn + c], acc);
    if (relu) acc = fmaxf(acc, 0.f);
    C[(long)m * Nn + c] = acc;
}

// ---------------------------------------------------------------------------
// Attention: one block per (query n, head h). qkv layout [N, 768].
// ---------------------------------------------------------------------------
__global__ void attn_kernel(const float* __restrict__ qkv, float* __restrict__ o) {
    __shared__ float sc[NNODES];
    __shared__ float qv[64];
    __shared__ float red[128];
    int n = blockIdx.x, h = blockIdx.y, tid = threadIdx.x;

    if (tid < 64) qv[tid] = qkv[n * 768 + h * 64 + tid];
    __syncthreads();

    for (int m = tid; m < NNODES; m += 128) {
        const float* kr = qkv + m * 768 + 256 + h * 64;
        float d = 0.f;
        for (int t = 0; t < 64; ++t) d = fmaf(qv[t], kr[t], d);
        sc[m] = d * 0.125f;
    }
    __syncthreads();

    float mx = -1e30f;
    for (int m = tid; m < NNODES; m += 128) mx = fmaxf(mx, sc[m]);
    red[tid] = mx; __syncthreads();
    for (int s = 64; s > 0; s >>= 1) {
        if (tid < s) red[tid] = fmaxf(red[tid], red[tid + s]);
        __syncthreads();
    }
    mx = red[0];
    __syncthreads();

    float ls = 0.f;
    for (int m = tid; m < NNODES; m += 128) {
        float e = __expf(sc[m] - mx);
        sc[m] = e;
        ls += e;
    }
    red[tid] = ls; __syncthreads();
    for (int s = 64; s > 0; s >>= 1) {
        if (tid < s) red[tid] += red[tid + s];
        __syncthreads();
    }
    float inv = 1.f / red[0];

    if (tid < 64) {
        float acc = 0.f;
        for (int m = 0; m < NNODES; ++m)
            acc = fmaf(sc[m], qkv[m * 768 + 512 + h * 64 + tid], acc);
        o[n * DMODEL + h * 64 + tid] = acc * inv;
    }
}

// ---------------------------------------------------------------------------
// x[r] = LayerNorm(x[r] + p[r]) * g + b   (in place, one block per row, D=256)
// ---------------------------------------------------------------------------
__global__ void addln_kernel(float* __restrict__ x, const float* __restrict__ p,
                             const float* __restrict__ g, const float* __restrict__ b) {
    __shared__ float red[DMODEL];
    int r = blockIdx.x, tid = threadIdx.x;
    float v = x[r * DMODEL + tid] + p[r * DMODEL + tid];
    red[tid] = v; __syncthreads();
    for (int s = 128; s > 0; s >>= 1) {
        if (tid < s) red[tid] += red[tid + s];
        __syncthreads();
    }
    float mu = red[0] * (1.f / DMODEL);
    __syncthreads();
    float d = v - mu;
    red[tid] = d * d; __syncthreads();
    for (int s = 128; s > 0; s >>= 1) {
        if (tid < s) red[tid] += red[tid + s];
        __syncthreads();
    }
    float rs = rsqrtf(red[0] * (1.f / DMODEL) + 1e-5f);
    x[r * DMODEL + tid] = d * rs * g[tid] + b[tid];
}

// ---------------------------------------------------------------------------
// Pair-MLP layer-1 decomposition, emitting f16:
//   z=0: Ah[r] = f16(x[r]@W1[0:256]   + boxes[r]@W1[512:516] + 0.5*b1)
//   z=1: Bh[r] = f16(x[r]@W1[256:512] + boxes[r]@W1[516:520] + 0.5*b1)
// ---------------------------------------------------------------------------
__global__ void pair_proj_kernel(const float* __restrict__ x,
                                 const float* __restrict__ boxes,
                                 const float* __restrict__ w1,
                                 const float* __restrict__ b1,
                                 _Float16* __restrict__ Ah, _Float16* __restrict__ Bh) {
    int c = blockIdx.x * 128 + threadIdx.x;   // 0..511
    int r = blockIdx.y;
    int z = blockIdx.z;
    float acc = 0.5f * b1[c];
    const float* xr = x + r * DMODEL;
    for (int k = 0; k < DMODEL; ++k)
        acc = fmaf(xr[k], w1[(z * DMODEL + k) * 512 + c], acc);
    const float* br = boxes + r * 4;
    for (int t = 0; t < 4; ++t)
        acc = fmaf(br[t], w1[(512 + z * 4 + t) * 512 + c], acc);
    _Float16* dst = z ? Bh : Ah;
    dst[r * 512 + c] = (_Float16)acc;
}

// ---------------------------------------------------------------------------
// Precompute W2 (f32 [512,56]) into WMMA B-fragment layout, f16, padded to 64
// cols: w2h[((w*16+kt)*32 + lane)*16 + t]; B layout: lane<16 K={0..7,16..23},
// lane>=16 K={8..15,24..31} (within k-tile), column n = w*16 + lane%16.
// ---------------------------------------------------------------------------
__global__ void w2frag_kernel(const float* __restrict__ w2, _Float16* __restrict__ w2h) {
    int id = blockIdx.x * 256 + threadIdx.x;       // 0..2047 = (w,kt,lane)
    if (id >= 2048) return;
    int l    = id & 31;
    int kt   = (id >> 5) & 15;
    int w    = id >> 9;
    int half = l >> 4, ln = l & 15;
    int n    = w * 16 + ln;
    for (int t = 0; t < 16; ++t) {
        int k = kt * 32 + (t < 8 ? half * 8 + t : 16 + half * 8 + (t - 8));
        float v = (n < 56) ? w2[k * 56 + n] : 0.f;
        w2h[id * 16 + t] = (_Float16)v;
    }
}

// ---------------------------------------------------------------------------
// Pair-MLP layer 2 hot path.  Grid (24 j-tiles, 384 i), 128 thr = 4 waves,
// wave w owns class columns [16w,16w+16).  Dual accumulation chains
// (k-tiles 0..7 -> acc0, 8..15 -> acc1) so consecutive WMMAs touch disjoint
// registers and the f16-WMMA 4-coexec hazard windows are filled with the
// other chain's fragment math instead of v_nops.
// ---------------------------------------------------------------------------
__global__ void __launch_bounds__(128)
pair_mlp_kernel(const _Float16* __restrict__ Ah, const _Float16* __restrict__ Bh,
                const _Float16* __restrict__ w2h, const float* __restrict__ b2,
                float* __restrict__ out_logits) {
    __shared__ __align__(16) _Float16 sA[512];
    __shared__ __align__(16) _Float16 sB[16 * SBSTRIDE];

    int jt = blockIdx.x, i = blockIdx.y;
    int j0 = jt * 16;
    int tid = threadIdx.x;

    // Stage A-row (1 KB) and 16 B-rows (16 KB) as 16-byte chunks.
    const v8h* Ag = (const v8h*)(Ah + i * 512);            // 64 chunks
    for (int c = tid; c < 64; c += 128)
        *(v8h*)(sA + c * 8) = Ag[c];
    const v8h* Bg = (const v8h*)Bh;                        // 64 chunks per row
    for (int c = tid; c < 1024; c += 128) {
        int m = c >> 6, ko = c & 63;
        *(v8h*)(sB + m * SBSTRIDE + ko * 8) = Bg[(j0 + m) * 64 + ko];
    }
    __syncthreads();

    int w    = tid >> 5;
    int l    = tid & 31;
    int half = l >> 4;
    int ln   = l & 15;
    int n    = w * 16 + ln;          // class column 0..63 (valid < 56)

    // W2 B-fragments for this wave's column tile (32-byte vector loads).
    const v16h* wf = (const v16h*)w2h + (w * 16) * 32 + l;  // + kt*32 per tile

    // Per-lane base pointers fold the half-dependent K split once:
    // lane<16 reads K {0..7,16..23}, lane>=16 reads K {8..15,24..31}.
    const _Float16* ap = sA + half * 8;
    const _Float16* bp = sB + ln * SBSTRIDE + half * 8;

    v8h zero = {};
    v8f acc0 = {}, acc1 = {};
#pragma unroll
    for (int kt = 0; kt < 8; ++kt) {
        // chain 0: k-tile kt (K base kt*32)
        v8h a0 = *(const v8h*)(ap + kt * 32);
        v8h a1 = *(const v8h*)(ap + kt * 32 + 16);
        v8h b0 = *(const v8h*)(bp + kt * 32);
        v8h b1 = *(const v8h*)(bp + kt * 32 + 16);
        v8h h0 = __builtin_elementwise_max(a0 + b0, zero);
        v8h h1 = __builtin_elementwise_max(a1 + b1, zero);
        v16h af0 = __builtin_shufflevector(h0, h1, 0, 1, 2, 3, 4, 5, 6, 7,
                                           8, 9, 10, 11, 12, 13, 14, 15);
        // chain 1: k-tile kt+8 (K base kt*32 + 256)
        v8h c0 = *(const v8h*)(ap + kt * 32 + 256);
        v8h c1 = *(const v8h*)(ap + kt * 32 + 256 + 16);
        v8h d0 = *(const v8h*)(bp + kt * 32 + 256);
        v8h d1 = *(const v8h*)(bp + kt * 32 + 256 + 16);
        v8h g0 = __builtin_elementwise_max(c0 + d0, zero);
        v8h g1 = __builtin_elementwise_max(c1 + d1, zero);
        v16h af1 = __builtin_shufflevector(g0, g1, 0, 1, 2, 3, 4, 5, 6, 7,
                                           8, 9, 10, 11, 12, 13, 14, 15);

        acc0 = __builtin_amdgcn_wmma_f32_16x16x32_f16(
            false, af0, false, wf[kt * 32], (short)0, acc0, false, false);
        acc1 = __builtin_amdgcn_wmma_f32_16x16x32_f16(
            false, af1, false, wf[(kt + 8) * 32], (short)0, acc1, false, false);
    }

    float bias = (n < 56) ? b2[n] : 0.f;
#pragma unroll
    for (int r = 0; r < 8; ++r) {
        int m = r + half * 8;
        int j = j0 + m;
        if (j == i || n >= 56) continue;
        int p = i * (NNODES - 1) + j - (j > i ? 1 : 0);
        out_logits[(long)p * 56 + n] = acc0[r] + acc1[r] + bias;
    }
}

// ---------------------------------------------------------------------------
// pair_idx: row-major (i,j), i != j, written as float values.
// ---------------------------------------------------------------------------
__global__ void pair_idx_kernel(float* __restrict__ out_idx) {
    int p = blockIdx.x * 256 + threadIdx.x;
    if (p >= NPAIRS) return;
    int i = p / (NNODES - 1);
    int r = p - i * (NNODES - 1);
    int j = r + (r >= i ? 1 : 0);
    out_idx[2 * p]     = (float)i;
    out_idx[2 * p + 1] = (float)j;
}

// ---------------------------------------------------------------------------
extern "C" void kernel_launch(void* const* d_in, const int* in_sizes, int n_in,
                              void* d_out, int out_size, void* d_ws, size_t ws_size,
                              hipStream_t stream) {
    const float* node_feats = (const float*)d_in[0];
    const float* boxes      = (const float*)d_in[1];
    const float* wi         = (const float*)d_in[2];
    const float* bi         = (const float*)d_in[3];
    const float* wo         = (const float*)d_in[4];
    const float* bo         = (const float*)d_in[5];
    const float* fw1        = (const float*)d_in[6];
    const float* fb1        = (const float*)d_in[7];
    const float* fw2        = (const float*)d_in[8];
    const float* fb2        = (const float*)d_in[9];
    const float* ln1g       = (const float*)d_in[10];
    const float* ln1b       = (const float*)d_in[11];
    const float* ln2g       = (const float*)d_in[12];
    const float* ln2b       = (const float*)d_in[13];
    const float* mlp_w1     = (const float*)d_in[14];
    const float* mlp_b1     = (const float*)d_in[15];
    const float* mlp_w2     = (const float*)d_in[16];
    const float* mlp_b2     = (const float*)d_in[17];

    float* ws  = (float*)d_ws;
    float* x   = ws;                          // 98304 floats
    float* qkv = ws + 98304;                  // 294912
    float* o   = ws + 393216;                 // 98304
    float* f1  = ws + 491520;                 // 393216
    _Float16* Ah  = (_Float16*)(ws + 884736); // 196608 halves
    _Float16* Bh  = (_Float16*)(ws + 983040); // 196608 halves
    _Float16* w2h = (_Float16*)(ws + 1081344);// 32768 halves

    hipMemcpyAsync(x, node_feats, (size_t)NNODES * DMODEL * sizeof(float),
                   hipMemcpyDeviceToDevice, stream);

    for (int lyr = 0; lyr < 2; ++lyr) {
        gemm_kernel<<<dim3(6, NNODES), 128, 0, stream>>>(
            x, wi + (long)lyr * 256 * 768, bi + lyr * 768, qkv, NNODES, 256, 768, 0);
        attn_kernel<<<dim3(NNODES, 4), 128, 0, stream>>>(qkv, o);
        gemm_kernel<<<dim3(2, NNODES), 128, 0, stream>>>(
            o, wo + (long)lyr * 256 * 256, bo + lyr * 256, qkv, NNODES, 256, 256, 0);
        addln_kernel<<<NNODES, 256, 0, stream>>>(x, qkv, ln1g + lyr * 256, ln1b + lyr * 256);
        gemm_kernel<<<dim3(8, NNODES), 128, 0, stream>>>(
            x, fw1 + (long)lyr * 256 * 1024, fb1 + lyr * 1024, f1, NNODES, 256, 1024, 1);
        gemm_kernel<<<dim3(2, NNODES), 128, 0, stream>>>(
            f1, fw2 + (long)lyr * 1024 * 256, fb2 + lyr * 256, qkv, NNODES, 1024, 256, 0);
        addln_kernel<<<NNODES, 256, 0, stream>>>(x, qkv, ln2g + lyr * 256, ln2b + lyr * 256);
    }

    pair_proj_kernel<<<dim3(4, NNODES, 2), 128, 0, stream>>>(
        x, boxes, mlp_w1, mlp_b1, Ah, Bh);
    w2frag_kernel<<<8, 256, 0, stream>>>(mlp_w2, w2h);

    float* out = (float*)d_out;
    pair_idx_kernel<<<(NPAIRS + 255) / 256, 256, 0, stream>>>(out);
    pair_mlp_kernel<<<dim3(24, NNODES), 128, 0, stream>>>(
        Ah, Bh, w2h, mlp_b2, out + 2 * (long)NPAIRS);
}
